// KMISPool_19215683683130
// MI455X (gfx1250) — matrix operand order repaired and checked by reading, not compiled, not run on hardware
//
#include <hip/hip_runtime.h>
#include <hip/hip_bf16.h>
#include <stdint.h>

typedef __attribute__((ext_vector_type(16))) _Float16 v16h;
typedef __attribute__((ext_vector_type(8)))  float    v8f;
typedef __attribute__((ext_vector_type(4)))  int      v4i;

#define TPB 256
static constexpr int MAX_LUBY = 24;
static constexpr int HBITS = 22;
static constexpr unsigned HSIZE = 1u << HBITS;
static constexpr unsigned long long HEMPTY = ~0ull;

// ---- CDNA5 async global->LDS copy (ASYNCcnt-tracked), with safe fallbacks ----
// Probe round 2 showed the builtin's first param is 'int __vector(4) __device__ *'
// (AS1 v4i*); cast both pointers to typed vector pointers in the right AS.
#if defined(__has_builtin)
#if __has_builtin(__builtin_amdgcn_global_load_async_to_lds_b128)
#define ASYNC_B128(gsrc, ldst)                                                  \
    __builtin_amdgcn_global_load_async_to_lds_b128(                             \
        (__attribute__((address_space(1))) v4i*)(gsrc),                         \
        (__attribute__((address_space(3))) v4i*)(ldst), 0, 0)
#define HAVE_ASYNC_LDS 1
#endif
#if __has_builtin(__builtin_amdgcn_s_wait_asynccnt)
#define WAIT_ASYNC() __builtin_amdgcn_s_wait_asynccnt(0)
#endif
#endif
#ifndef ASYNC_B128
#define ASYNC_B128(gsrc, ldst)                                                  \
    do { *(float4*)(ldst) = *(const float4*)(gsrc); } while (0)
#endif
#ifndef WAIT_ASYNC
#ifdef HAVE_ASYNC_LDS
#define WAIT_ASYNC() asm volatile("s_wait_asynccnt 0x0" ::: "memory")
#else
#define WAIT_ASYNC() do {} while (0)
#endif
#endif

// ---------------- small utility kernels ----------------
__global__ void k_fill_i32(int* p, int v, int n) {
    int i = blockIdx.x * blockDim.x + threadIdx.x;
    if (i < n) p[i] = v;
}
__global__ void k_fill_u64(unsigned long long* p, unsigned long long v, unsigned n) {
    unsigned i = blockIdx.x * blockDim.x + threadIdx.x;
    if (i < n) p[i] = v;
}
__global__ void k_iota(int* p, int n) {
    int i = blockIdx.x * blockDim.x + threadIdx.x;
    if (i < n) p[i] = i;
}
__global__ void k_copy_g(int* dst, const int* src, int n, const int* guard) {
    if (*guard == 0) return;
    int i = blockIdx.x * blockDim.x + threadIdx.x;
    if (i < n) dst[i] = src[i];
}

// ---------------- edge scatter kernels (guarded) ----------------
__global__ void k_edge_min(const int* __restrict__ row, const int* __restrict__ col,
                           const int* __restrict__ src, int* __restrict__ dst,
                           int e, const int* guard) {
    if (*guard == 0) return;
    int i = blockIdx.x * blockDim.x + threadIdx.x;
    if (i >= e) return;
    __builtin_prefetch((const void*)(row + i + 2048), 0, 1);  // global_prefetch_b8
    __builtin_prefetch((const void*)(col + i + 2048), 0, 1);
    atomicMin(&dst[col[i]], src[row[i]]);
}
__global__ void k_edge_or(const int* __restrict__ row, const int* __restrict__ col,
                          const int* __restrict__ src, int* __restrict__ dst,
                          int e, const int* guard) {
    if (*guard == 0) return;
    int i = blockIdx.x * blockDim.x + threadIdx.x;
    if (i >= e) return;
    __builtin_prefetch((const void*)(row + i + 2048), 0, 1);
    if (src[row[i]]) atomicOr(&dst[col[i]], 1);
}

// ---------------- Luby iteration node kernels ----------------
__global__ void k_mis_update(int* mis, const int* mr, int n, const int* guard) {
    if (*guard == 0) return;
    int i = blockIdx.x * blockDim.x + threadIdx.x;
    if (i < n && mr[i] == i) mis[i] = 1;
}
__global__ void k_rank_reset(int* mr, const int* mask, int nn, int* uncovered, const int* guard) {
    if (*guard == 0) return;
    int i = blockIdx.x * blockDim.x + threadIdx.x;
    if (i >= nn) return;
    if (mask[i]) { mr[i] = nn; }
    else         { mr[i] = i; atomicOr(uncovered, 1); }
}

// ---------------- clustering kernels ----------------
__global__ void k_cluster_init(int* mr, const int* mis, int nn) {
    int i = blockIdx.x * blockDim.x + threadIdx.x;
    if (i < nn) mr[i] = mis[i] ? i : nn;
}
__global__ void k_mark(const int* mr, int* pres, int n) {
    int i = blockIdx.x * blockDim.x + threadIdx.x;
    if (i < n) pres[mr[i]] = 1;
}
__global__ void k_gather_prefix(int* cluster, const int* mr, const int* prefix, int n) {
    int i = blockIdx.x * blockDim.x + threadIdx.x;
    if (i < n) cluster[i] = prefix[mr[i]];
}
__global__ void k_count(const int* cluster, int* counts, int n) {
    int i = blockIdx.x * blockDim.x + threadIdx.x;
    if (i < n) atomicAdd(&counts[cluster[i]], 1);
}
__global__ void k_scatter(const int* cluster, int* cursor, int* sorted, int n) {
    int i = blockIdx.x * blockDim.x + threadIdx.x;
    if (i < n) { int p = atomicAdd(&cursor[cluster[i]], 1); sorted[p] = i; }
}

// ---------------- exclusive scan (3-kernel, deterministic) ----------------
__global__ void k_scan_block(const int* in, int* out, int* bsums, int n) {
    __shared__ int tmp[1024];
    int gid = blockIdx.x * 1024 + threadIdx.x;
    int v = (gid < n) ? in[gid] : 0;
    tmp[threadIdx.x] = v;
    __syncthreads();
    for (int s = 1; s < 1024; s <<= 1) {
        int t = (threadIdx.x >= s) ? tmp[threadIdx.x - s] : 0;
        __syncthreads();
        tmp[threadIdx.x] += t;
        __syncthreads();
    }
    if (gid < n) out[gid] = tmp[threadIdx.x] - v;
    if (threadIdx.x == 1023) bsums[blockIdx.x] = tmp[1023];
}
__global__ void k_scan_single(int* data, int n, int* total) {
    __shared__ int tmp[1024];
    __shared__ int carry;
    if (threadIdx.x == 0) carry = 0;
    __syncthreads();
    for (int base = 0; base < n; base += 1024) {
        int gid = base + threadIdx.x;
        int v = (gid < n) ? data[gid] : 0;
        tmp[threadIdx.x] = v;
        __syncthreads();
        for (int s = 1; s < 1024; s <<= 1) {
            int t = (threadIdx.x >= s) ? tmp[threadIdx.x - s] : 0;
            __syncthreads();
            tmp[threadIdx.x] += t;
            __syncthreads();
        }
        if (gid < n) data[gid] = carry + tmp[threadIdx.x] - v;
        __syncthreads();
        if (threadIdx.x == 0) carry += tmp[1023];
        __syncthreads();
    }
    if (threadIdx.x == 0 && total) *total = carry;
}
__global__ void k_scan_add(int* out, const int* bsums, int n) {
    int gid = blockIdx.x * 1024 + threadIdx.x;
    if (gid < n) out[gid] += bsums[blockIdx.x];
}

// ---------------- WMMA segment-sum: x_pooled = S^T x ----------------
// One block per 16 consecutive cluster ids; 4 waves x 4 col-blocks cover D=256.
// Per 32-node chunk: async-stage 32x256 f32 features to LDS (b128, ASYNCcnt),
// A = one-hot(16x32 f16) from LDS cluster-locals, B = f16-converted LDS reads,
// acc 16x16 f32 via v_wmma_f32_16x16x32_f16.
__global__ __launch_bounds__(128)
void k_pool_x_wmma(const float* __restrict__ x, const int* __restrict__ sorted,
                   const int* __restrict__ offsets, const int* __restrict__ cluster,
                   const int* __restrict__ scalars, float* __restrict__ out) {
    __shared__ float s_x[32 * 256];   // 32 KB feature staging (node-major)
    __shared__ int   s_cl[32];

    int c  = scalars[0];
    int gb = blockIdx.x * 16;
    if (gb >= c) return;                       // uniform exit for whole block
    int lane = threadIdx.x & 31;
    int wave = threadIdx.x >> 5;

    int hi  = gb + 16; if (hi > c) hi = c;
    int start = offsets[gb];
    int end   = offsets[hi];

    v8f acc0 = {}, acc1 = {}, acc2 = {}, acc3 = {};
    const int m = lane & 15;
    const int half = lane >> 4;
    const int ncol = lane & 15;
    const int kbase = half * 16;

    for (int base = start; base < end; base += 32) {
        __syncthreads();   // LDS consumed by previous iteration is done

        // stage cluster-locals for the chunk's 32 nodes (threads 0..31)
        if (threadIdx.x < 32) {
            int cl = 255;
            if (base + threadIdx.x < end) {
                int nd = sorted[base + threadIdx.x];
                cl = cluster[nd] - gb;
            }
            s_cl[threadIdx.x] = cl;
        }
        __syncthreads();

        // stage features: 32 nodes x 256 f32 = 2048 float4, 16 per thread,
        // coalesced b128 rows via async global->LDS (zero-fill padded rows)
        for (int q = 0; q < 16; ++q) {
            int l = q * 128 + threadIdx.x;      // float4 slot
            int ni = l >> 6;                    // node within chunk
            int f4 = l & 63;                    // float4 within row
            int valid = (base + ni < end);
            int nd = valid ? sorted[base + ni] : -1;
            float* ldst = &s_x[ni * 256 + f4 * 4];
            if (nd >= 0) {
                ASYNC_B128(x + (size_t)nd * 256 + (size_t)f4 * 4, ldst);
            } else {
                *(float4*)ldst = make_float4(0.f, 0.f, 0.f, 0.f);
            }
        }
        WAIT_ASYNC();        // this wave's async copies complete
        __syncthreads();     // all waves' copies visible

        // ---- A: one-hot, 16-bit A-matrix 16x32 layout (ISA 7.12.2) ----
        v16h A;
#pragma unroll
        for (int j = 0; j < 8; ++j) {
            int k0 = (j < 4) ? (half * 8 + 2 * j) : (16 + half * 8 + 2 * (j - 4));
            A[2 * j]     = (_Float16)((s_cl[k0]     == m) ? 1.0f : 0.0f);
            A[2 * j + 1] = (_Float16)((s_cl[k0 + 1] == m) ? 1.0f : 0.0f);
        }

        // ---- B: lanes 0-15 K=0..7,16..23, lanes 16-31 K=8..15,24..31 ----
#pragma unroll
        for (int fb = 0; fb < 4; ++fb) {
            int feat = (wave * 4 + fb) * 16 + ncol;
            v16h B;
#pragma unroll
            for (int e = 0; e < 16; ++e)
                B[e] = (_Float16)s_x[(half * 8 + e + ((e >= 8) ? 8 : 0)) * 256 + feat];
            v8f a = (fb == 0) ? acc0 : (fb == 1) ? acc1 : (fb == 2) ? acc2 : acc3;
            a = __builtin_amdgcn_wmma_f32_16x16x32_f16(false, A, false, B,
                                                       (short)0, a, false, false);
            if (fb == 0) acc0 = a; else if (fb == 1) acc1 = a;
            else if (fb == 2) acc2 = a; else acc3 = a;
        }
    }

    // C/D layout: VGPR r holds M = r + 8*half, N = lane&15
#pragma unroll
    for (int fb = 0; fb < 4; ++fb) {
        v8f a = (fb == 0) ? acc0 : (fb == 1) ? acc1 : (fb == 2) ? acc2 : acc3;
        int feat = (wave * 4 + fb) * 16 + ncol;
#pragma unroll
        for (int r = 0; r < 8; ++r) {
            int crow = gb + r + 8 * half;
            if (crow < c) out[(size_t)crow * 256 + feat] = a[r];
        }
    }
}

// ---------------- pooled-edge coalescing via hash ----------------
__global__ void k_edge_hash(const int* __restrict__ row, const int* __restrict__ col,
                            const float* __restrict__ attr, const int* __restrict__ cluster,
                            const int* __restrict__ scalars,
                            unsigned long long* __restrict__ keys, float* __restrict__ vals,
                            int e) {
    int i = blockIdx.x * blockDim.x + threadIdx.x;
    if (i >= e) return;
    int c = scalars[0];
    if (c <= 0) return;
    __builtin_prefetch((const void*)(attr + i + 2048), 0, 1);
    int cr = cluster[row[i]], cc = cluster[col[i]];
    if (cr == cc) return;                                  // remove_self_loops
    unsigned long long key = (unsigned long long)cr * (unsigned)c + (unsigned)cc;
    unsigned long long h = key * 0x9E3779B97F4A7C15ull;
    unsigned slot = (unsigned)(h >> (64 - HBITS));
    for (;;) {
        unsigned long long cur = keys[slot];
        if (cur == key) break;
        if (cur == HEMPTY) {
            unsigned long long prev = atomicCAS(&keys[slot], HEMPTY, key);
            if (prev == HEMPTY || prev == key) break;
        }
        slot = (slot + 1) & (HSIZE - 1);
    }
    atomicAdd(&vals[slot], attr[i]);
}
__global__ void k_hflag(const unsigned long long* keys, int* flag, unsigned n) {
    unsigned i = blockIdx.x * blockDim.x + threadIdx.x;
    if (i < n) flag[i] = (keys[i] != HEMPTY) ? 1 : 0;
}
__global__ void k_hash_emit(const unsigned long long* __restrict__ keys,
                            const float* __restrict__ vals, const int* __restrict__ pos,
                            const int* __restrict__ scalars, float* __restrict__ out,
                            unsigned n) {
    unsigned i = blockIdx.x * blockDim.x + threadIdx.x;
    if (i >= n) return;
    unsigned long long k = keys[i];
    if (k == HEMPTY) return;
    int c = scalars[0];
    if (c <= 0) return;
    int ecnt = scalars[1];
    long long base = (long long)c * 256;
    int idx = pos[i];
    unsigned long long r  = k / (unsigned)c;
    unsigned long long cc = k % (unsigned)c;
    out[base + idx]            = (float)r;   // pooled_edge_index row
    out[base + ecnt + idx]     = (float)cc;  // pooled_edge_index col
    out[base + 2 * ecnt + idx] = vals[i];    // pooled_edge_attr
}

// ---------------- host orchestration ----------------
static void scan_exclusive(const int* in, int* out, int n, int* bsums, int* total,
                           hipStream_t s) {
    int nb = (n + 1023) / 1024;
    k_scan_block<<<nb, 1024, 0, s>>>(in, out, bsums, n);
    k_scan_single<<<1, 1024, 0, s>>>(bsums, nb, total);
    k_scan_add<<<nb, 1024, 0, s>>>(out, bsums, n);
}

extern "C" void kernel_launch(void* const* d_in, const int* in_sizes, int n_in,
                              void* d_out, int out_size, void* d_ws, size_t ws_size,
                              hipStream_t stream) {
    (void)n_in; (void)out_size; (void)ws_size;
    const float* x  = (const float*)d_in[0];
    const int*   ei = (const int*)d_in[1];
    const float* ea = (const float*)d_in[2];
    const int N = in_sizes[0] / 256;
    const int E = in_sizes[2];
    const int* row = ei;
    const int* col = ei + E;
    float* out = (float*)d_out;

    char* w = (char*)d_ws; size_t off = 0;
    auto alloci = [&](size_t ne) -> int* {
        int* p = (int*)(w + off); off += ne * 4; off = (off + 255) & ~(size_t)255; return p;
    };
    int* mr_a    = alloci(N);
    int* mr_b    = alloci(N);
    int* mis     = alloci(N);
    int* mask_a  = alloci(N);
    int* mask_b  = alloci(N);
    int* cluster = alloci(N);
    int* sorted  = alloci(N);
    int* presence = alloci(N + 2);
    int* prefix   = alloci(N + 2);
    int* counts   = alloci(N + 2);
    int* offsets  = alloci(N + 2);
    int* cursor   = alloci(N + 2);
    int* flags    = alloci(MAX_LUBY + 2);
    int* scalars  = alloci(16);      // [0]=c, [1]=E', [2]=const 1
    int* bsums    = alloci(8192);
    unsigned long long* hkeys = (unsigned long long*)(w + off); off += (size_t)HSIZE * 8;
    float* hvals = (float*)(w + off); off += (size_t)HSIZE * 4;
    int* hflag = alloci(HSIZE);
    int* hpos  = alloci(HSIZE);

    const int nbN  = (N + TPB - 1) / TPB;
    const int nbN1 = (N + 2 + TPB - 1) / TPB;
    const int nbE  = (E + TPB - 1) / TPB;
    const int nbH  = (int)((HSIZE + TPB - 1) / TPB);

    // init
    k_iota<<<nbN, TPB, 0, stream>>>(mr_a, N);
    k_fill_i32<<<nbN, TPB, 0, stream>>>(mis, 0, N);
    k_fill_i32<<<nbN, TPB, 0, stream>>>(mask_a, 0, N);
    k_fill_i32<<<1, 64, 0, stream>>>(flags, 0, MAX_LUBY + 2);
    k_fill_i32<<<1, 64, 0, stream>>>(scalars, 0, 16);
    k_fill_i32<<<1, 32, 0, stream>>>(flags, 1, 1);        // flags[0] = 1: run iter 0
    k_fill_i32<<<1, 32, 0, stream>>>(scalars + 2, 1, 1);  // const one
    const int* ones = scalars + 2;

    // Luby-style greedy k-MIS (k = 2), fixed launch count, device-side guards
    int* cur = mr_a; int* nxt = mr_b;
    int* mcur = mask_a; int* mnxt = mask_b;
    for (int it = 0; it < MAX_LUBY; ++it) {
        const int* g = flags + it;
        int* unc = flags + it + 1;
        for (int h = 0; h < 2; ++h) {   // propagate_min, k hops
            k_copy_g<<<nbN, TPB, 0, stream>>>(nxt, cur, N, g);
            k_edge_min<<<nbE, TPB, 0, stream>>>(row, col, cur, nxt, E, g);
            int* t = cur; cur = nxt; nxt = t;
        }
        k_mis_update<<<nbN, TPB, 0, stream>>>(mis, cur, N, g);
        k_copy_g<<<nbN, TPB, 0, stream>>>(mcur, mis, N, g);
        for (int h = 0; h < 2; ++h) {   // dilate, k hops
            k_copy_g<<<nbN, TPB, 0, stream>>>(mnxt, mcur, N, g);
            k_edge_or<<<nbE, TPB, 0, stream>>>(row, col, mcur, mnxt, E, g);
            int* t = mcur; mcur = mnxt; mnxt = t;
        }
        k_rank_reset<<<nbN, TPB, 0, stream>>>(cur, mcur, N, unc, g);
    }

    // cluster assignment: min-rank from MIS, 2 hops
    k_cluster_init<<<nbN, TPB, 0, stream>>>(cur, mis, N);
    for (int h = 0; h < 2; ++h) {
        k_copy_g<<<nbN, TPB, 0, stream>>>(nxt, cur, N, ones);
        k_edge_min<<<nbE, TPB, 0, stream>>>(row, col, cur, nxt, E, ones);
        int* t = cur; cur = nxt; nxt = t;
    }
    // unique(min_rank) -> dense cluster ids via presence bitmap + scan
    k_fill_i32<<<nbN1, TPB, 0, stream>>>(presence, 0, N + 2);
    k_mark<<<nbN, TPB, 0, stream>>>(cur, presence, N);
    scan_exclusive(presence, prefix, N + 1, bsums, scalars + 0, stream);  // scalars[0]=c
    k_gather_prefix<<<nbN, TPB, 0, stream>>>(cluster, cur, prefix, N);

    // counting sort of nodes by cluster
    k_fill_i32<<<nbN1, TPB, 0, stream>>>(counts, 0, N + 2);
    k_count<<<nbN, TPB, 0, stream>>>(cluster, counts, N);
    scan_exclusive(counts, offsets, N + 1, bsums, nullptr, stream);
    k_copy_g<<<nbN1, TPB, 0, stream>>>(cursor, offsets, N + 1, ones);
    k_scatter<<<nbN, TPB, 0, stream>>>(cluster, cursor, sorted, N);

    // pooled features via WMMA (one block per 16-cluster group)
    k_pool_x_wmma<<<(N + 15) / 16, 128, 0, stream>>>(x, sorted, offsets, cluster,
                                                     scalars, out);

    // pooled adjacency: hash-coalesce, compact, emit
    k_fill_u64<<<nbH, TPB, 0, stream>>>(hkeys, HEMPTY, HSIZE);
    k_fill_i32<<<nbH, TPB, 0, stream>>>((int*)hvals, 0, HSIZE);
    k_edge_hash<<<nbE, TPB, 0, stream>>>(row, col, ea, cluster, scalars, hkeys, hvals, E);
    k_hflag<<<nbH, TPB, 0, stream>>>(hkeys, hflag, HSIZE);
    scan_exclusive(hflag, hpos, (int)HSIZE, bsums, scalars + 1, stream);  // scalars[1]=E'
    k_hash_emit<<<nbH, TPB, 0, stream>>>(hkeys, hvals, hpos, scalars, out, HSIZE);
}
